// rnnAttn_26396869001582
// MI455X (gfx1250) — matrix operand order repaired and checked by reading, not compiled, not run on hardware
//
#include <hip/hip_runtime.h>
#include <hip/hip_bf16.h>
#include <math.h>

typedef __attribute__((ext_vector_type(16))) _Float16 v16h;
typedef __attribute__((ext_vector_type(8)))  _Float16 v8h;
typedef __attribute__((ext_vector_type(8)))  float    v8f;

#define WMMA(a,b,c) __builtin_amdgcn_wmma_f32_16x16x32_f16(false,(a),false,(b),(short)0,(c),false,false)

// B=32, L=1024, D=256, G=768, M=B*L=32768

// ---------------------------------------------------------------------------
// Packing kernels.
// A-pack (16x32 tiles, rows=M, K): halfidx = ((mt*KT+kt)*32+lane)*16+elem
//   m = mt*16+(lane&15); k = kt*32+(elem&7)+16*((elem>>3)&1)+8*(lane>>4)
// B-pack (32x16 tiles, K x N):     halfidx = ((nt*KT+kt)*32+lane)*16+elem
//   n = nt*16+(lane&15); k = kt*32+(elem&7)+16*((elem>>3)&1)+8*(lane>>4)
// ---------------------------------------------------------------------------
__global__ void pack_a_kernel(const float* __restrict__ X, _Float16* __restrict__ out,
                              int R, int K) {
  size_t i = (size_t)blockIdx.x * blockDim.x + threadIdx.x;
  size_t total = (size_t)R * K;
  if (i >= total) return;
  int elem = i & 15, lane = (i >> 4) & 31;
  size_t tile = i >> 9;
  int KT = K >> 5;
  int kt = (int)(tile % KT), mt = (int)(tile / KT);
  int m = mt * 16 + (lane & 15);
  int k = kt * 32 + (elem & 7) + 16 * ((elem >> 3) & 1) + 8 * (lane >> 4);
  out[i] = (_Float16)X[(size_t)m * K + k];
}

// X row-major [N][K] (weights W[n][k], or row-major activations used as B^T)
__global__ void pack_b_kernel(const float* __restrict__ X, _Float16* __restrict__ out,
                              int N, int K) {
  X   += (size_t)blockIdx.y * N * K;
  out += (size_t)blockIdx.y * N * K;
  size_t i = (size_t)blockIdx.x * blockDim.x + threadIdx.x;
  size_t total = (size_t)N * K;
  if (i >= total) return;
  int elem = i & 15, lane = (i >> 4) & 31;
  size_t tile = i >> 9;
  int KT = K >> 5;
  int kt = (int)(tile % KT), nt = (int)(tile / KT);
  int n = nt * 16 + (lane & 15);
  int k = kt * 32 + (elem & 7) + 16 * ((elem >> 3) & 1) + 8 * (lane >> 4);
  out[i] = (_Float16)X[(size_t)n * K + k];
}

// X row-major [K][N] (e.g. V matrix: outs[m][d] with K=m)
__global__ void pack_bT_kernel(const float* __restrict__ X, _Float16* __restrict__ out,
                               int N, int K) {
  X   += (size_t)blockIdx.y * N * K;
  out += (size_t)blockIdx.y * N * K;
  size_t i = (size_t)blockIdx.x * blockDim.x + threadIdx.x;
  size_t total = (size_t)N * K;
  if (i >= total) return;
  int elem = i & 15, lane = (i >> 4) & 31;
  size_t tile = i >> 9;
  int KT = K >> 5;
  int kt = (int)(tile % KT), nt = (int)(tile / KT);
  int n = nt * 16 + (lane & 15);
  int k = kt * 32 + (elem & 7) + 16 * ((elem >> 3) & 1) + 8 * (lane >> 4);
  out[i] = (_Float16)X[(size_t)k * N + n];
}

// broadcast sh[b][k] into one row-tile of A-layout per (b,kt):  32*8*512 halfs
__global__ void shrep_kernel(const float* __restrict__ sh, _Float16* __restrict__ out) {
  int i = blockIdx.x * blockDim.x + threadIdx.x;
  if (i >= 32 * 8 * 512) return;
  int elem = i & 15, lane = (i >> 4) & 31;
  int tile = i >> 9;
  int kt = tile & 7, b = tile >> 3;
  int k = kt * 32 + (elem & 7) + 16 * ((elem >> 3) & 1) + 8 * (lane >> 4);
  out[i] = (_Float16)sh[b * 256 + k];
}

// ---------------------------------------------------------------------------
// Generic WMMA GEMM: C[M][N] f32 = Apack(MxK) x Bpack(KxN) + bias[N]
// one 16x16 tile per wave, 8 waves per block
// ---------------------------------------------------------------------------
__global__ __launch_bounds__(256)
void gemm_kernel(const v16h* __restrict__ Ap, const v16h* __restrict__ Bp,
                 const float* __restrict__ bias, float* __restrict__ C,
                 int MT, int NT, int KT) {
  int wv = (int)((blockIdx.x * blockDim.x + threadIdx.x) >> 5);
  int lane = threadIdx.x & 31;
  if (wv >= MT * NT) return;
  int mt = wv / NT, nt = wv % NT;
  const v16h* a = Ap + (size_t)mt * KT * 32 + lane;
  const v16h* b = Bp + (size_t)nt * KT * 32 + lane;
  v8f acc = {};
  for (int kt = 0; kt < KT; ++kt)
    acc = WMMA(a[kt * 32], b[kt * 32], acc);
  int N = NT * 16;
  int n = nt * 16 + (lane & 15);
  float bv = bias ? bias[n] : 0.f;
  int hi = lane >> 4;
#pragma unroll
  for (int r = 0; r < 8; ++r) {
    int m = mt * 16 + r + 8 * hi;
    C[(size_t)m * N + n] = acc[r] + bv;
  }
}

// comb GEMM: K=768 split into ctx(0:256) | outs(256:512) | sh(512:768)
__global__ __launch_bounds__(256)
void gemm_comb_kernel(const v16h* __restrict__ ApCtx, const v16h* __restrict__ ApOuts,
                      const v16h* __restrict__ ApSh, const v16h* __restrict__ Bp,
                      const float* __restrict__ bias, float* __restrict__ C) {
  const int MT = 2048, NT = 16;
  int wv = (int)((blockIdx.x * blockDim.x + threadIdx.x) >> 5);
  int lane = threadIdx.x & 31;
  if (wv >= MT * NT) return;
  int mt = wv / NT, nt = wv % NT;
  int b = mt >> 6;
  v8f acc = {};
  for (int kt = 0; kt < 24; ++kt) {
    v16h a;
    if (kt < 8)       a = ApCtx [((size_t)mt * 8 + kt) * 32 + lane];
    else if (kt < 16) a = ApOuts[((size_t)mt * 8 + (kt - 8)) * 32 + lane];
    else              a = ApSh  [((size_t)(b * 8 + (kt - 16))) * 32 + lane];
    acc = WMMA(a, Bp[((size_t)nt * 24 + kt) * 32 + lane], acc);
  }
  int n = nt * 16 + (lane & 15);
  float bv = bias[n];
  int hi = lane >> 4;
#pragma unroll
  for (int r = 0; r < 8; ++r) {
    int m = mt * 16 + r + 8 * hi;
    C[(size_t)m * 256 + n] = acc[r] + bv;
  }
}

// ---------------------------------------------------------------------------
// Elementwise
// ---------------------------------------------------------------------------
__global__ void gelu_kernel(const float* __restrict__ x, float* __restrict__ y, int n) {
  int i = blockIdx.x * blockDim.x + threadIdx.x;
  if (i < n) { float v = x[i]; y[i] = 0.5f * v * (1.f + erff(v * 0.70710678118654752f)); }
}
__global__ void tanh_kernel(float* __restrict__ y, int n) {
  int i = blockIdx.x * blockDim.x + threadIdx.x;
  if (i < n) y[i] = tanhf(y[i]);
}
__global__ void add_kernel(const float* __restrict__ a, const float* __restrict__ b,
                           float* __restrict__ c, int n) {
  int i = blockIdx.x * blockDim.x + threadIdx.x;
  if (i < n) c[i] = a[i] + b[i];
}

// ---------------------------------------------------------------------------
// Persistent GRU layer: one workgroup (32 waves), 1024 steps.
// Wave wv: mt=wv>>4 (batch half), nt0=wv&15 (d-column tile). It computes the
// r/z/n gate tiles (nt0, nt0+16, nt0+32) so gate math is register-local.
// h kept in registers (hreg) + A-layout f16 copy in LDS (hA) for WMMA.
// Layer0: x-gate from scalar trg path. Layer1: x-gate GEMM from xAp (o0).
// ---------------------------------------------------------------------------
__global__ __launch_bounds__(1024)
void gru_kernel(const float* __restrict__ h0,
                const v16h* __restrict__ BpWhh, const float* __restrict__ b_hh,
                const float* __restrict__ b_ih,
                const int* __restrict__ trg_seq, const float* __restrict__ w_ih0,
                const v16h* __restrict__ xAp, const v16h* __restrict__ BpWih,
                _Float16* __restrict__ outAp, float* __restrict__ outF32) {
  __shared__ __align__(32) _Float16 hA[8192];   // 2 mt x 8 kt x 32 lanes x 16
  int tid = threadIdx.x, lane = tid & 31, wv = tid >> 5;
  int hi = lane >> 4, col = lane & 15;
  int mt = wv >> 4, nt0 = wv & 15;

  for (int i = tid; i < 8192; i += 1024) {
    int el = i & 15, ln = (i >> 4) & 31, t = i >> 9;
    int kt = t & 7, mtile = t >> 3;
    int bb = mtile * 16 + (ln & 15);
    int k = kt * 32 + (el & 7) + 16 * ((el >> 3) & 1) + 8 * (ln >> 4);
    hA[i] = (_Float16)h0[bb * 256 + k];
  }
  int d = nt0 * 16 + col;
  float hreg[8];
#pragma unroll
  for (int r = 0; r < 8; ++r) hreg[r] = h0[(mt * 16 + r + 8 * hi) * 256 + d];
  float bhr = b_hh[d], bhz = b_hh[256 + d], bhn = b_hh[512 + d];
  float bir = b_ih[d], biz = b_ih[256 + d], bin_ = b_ih[512 + d];
  float wir = 0.f, wiz = 0.f, win = 0.f;
  if (trg_seq) { wir = w_ih0[d]; wiz = w_ih0[256 + d]; win = w_ih0[512 + d]; }
  __syncthreads();

  for (int l = 0; l < 1024; ++l) {
    v8f aR = {}, aZ = {}, aN = {};
    const v16h* Ah = ((const v16h*)hA) + (size_t)mt * 8 * 32 + lane;
#pragma unroll
    for (int kt = 0; kt < 8; ++kt) {
      v16h a = Ah[kt * 32];
      aR = WMMA(a, BpWhh[((size_t)nt0 * 8 + kt) * 32 + lane], aR);
      aZ = WMMA(a, BpWhh[((size_t)(nt0 + 16) * 8 + kt) * 32 + lane], aZ);
      aN = WMMA(a, BpWhh[((size_t)(nt0 + 32) * 8 + kt) * 32 + lane], aN);
    }
    v8f xR = {}, xZ = {}, xN = {};
    if (xAp) {
      const v16h* Ax = xAp + (size_t)(l * 2 + mt) * 8 * 32 + lane;
#pragma unroll
      for (int kt = 0; kt < 8; ++kt) {
        v16h a = Ax[kt * 32];
        xR = WMMA(a, BpWih[((size_t)nt0 * 8 + kt) * 32 + lane], xR);
        xZ = WMMA(a, BpWih[((size_t)(nt0 + 16) * 8 + kt) * 32 + lane], xZ);
        xN = WMMA(a, BpWih[((size_t)(nt0 + 32) * 8 + kt) * 32 + lane], xN);
      }
    }
    __syncthreads();  // all hA reads done before updates
#pragma unroll
    for (int r = 0; r < 8; ++r) {
      int b = mt * 16 + r + 8 * hi;
      float xr, xz, xn;
      if (trg_seq) {
        float xin = (l == 0) ? 1.0f : (float)trg_seq[b * 1024 + l - 1];
        xr = xin * wir + bir; xz = xin * wiz + biz; xn = xin * win + bin_;
      } else {
        xr = xR[r] + bir; xz = xZ[r] + biz; xn = xN[r] + bin_;
      }
      float hr = aR[r] + bhr, hz = aZ[r] + bhz, hn = aN[r] + bhn;
      float rr = 1.f / (1.f + expf(-(xr + hr)));
      float zz = 1.f / (1.f + expf(-(xz + hz)));
      float nn = tanhf(xn + rr * hn);
      float hv = (1.f - zz) * nn + zz * hreg[r];
      hreg[r] = hv;
      int kt = d >> 5, kk = d & 31;
      int ln = (b & 15) + 16 * ((kk >> 3) & 1);
      int el = (kk & 7) + 8 * ((kk >> 4) & 1);
      hA[(((b >> 4) * 8 + kt) * 32 + ln) * 16 + el] = (_Float16)hv;
      if (outAp) {  // rows m = l*32+b, A-pack, KT=8
        int m = l * 32 + b;
        outAp[(((size_t)(m >> 4) * 8 + kt) * 32 + ((m & 15) + 16 * ((kk >> 3) & 1))) * 16 + el]
            = (_Float16)hv;
      }
      if (outF32) outF32[((size_t)b * 1024 + l) * 256 + d] = hv;  // b-major
    }
    __syncthreads();
  }
}

// ---------------------------------------------------------------------------
// Flash-style causal attention: one wave per (b, 16-query tile).
// S = Q x modif^T via WMMA; online softmax; P x V via WMMA over 32-key pairs.
// Diagonal excluded (m < l); fully-masked rows (l==0) -> ctx = 0.
// ---------------------------------------------------------------------------
__global__ __launch_bounds__(32)
void attn_kernel(const v16h* __restrict__ ApQ, const v16h* __restrict__ BpM,
                 const v16h* __restrict__ BpV, float* __restrict__ ctx) {
  __shared__ __align__(32) _Float16 pf[16 * 32];
  int lane = threadIdx.x;
  int hi = lane >> 4, col = lane & 15;
  int b = blockIdx.x >> 6, lt = blockIdx.x & 63;
  size_t mtile = (size_t)b * 64 + lt;
  v16h q[8];
#pragma unroll
  for (int kt = 0; kt < 8; ++kt) q[kt] = ApQ[(mtile * 8 + kt) * 32 + lane];
  v8f acc[16] = {};
  float mx[8], sm[8];
#pragma unroll
  for (int r = 0; r < 8; ++r) { mx[r] = -3.0e38f; sm[r] = 0.f; }
  const v16h* BMb = BpM + (size_t)b * (64 * 8 * 32);
  const v16h* BVb = BpV + (size_t)b * (16 * 32 * 32);
  int npairs = (lt >> 1) + 1;
  for (int kp = 0; kp < npairs; ++kp) {
    int jt0 = 2 * kp, jt1 = 2 * kp + 1;
    v8f s0 = {}, s1 = {};
    const v16h* B0 = BMb + (size_t)jt0 * 8 * 32 + lane;
    const v16h* B1 = B0 + 8 * 32;
#pragma unroll
    for (int kt = 0; kt < 8; ++kt) {
      s0 = WMMA(q[kt], B0[kt * 32], s0);
      s1 = WMMA(q[kt], B1[kt * 32], s1);
    }
#pragma unroll
    for (int r = 0; r < 8; ++r) {
      int mrow = r + 8 * hi;
      float v0 = ((jt0 < lt) || (jt0 == lt && col < mrow)) ? s0[r] : -3.0e38f;
      float v1 = ((jt1 < lt) || (jt1 == lt && col < mrow)) ? s1[r] : -3.0e38f;
      float rmax = fmaxf(v0, v1);
      rmax = fmaxf(rmax, __shfl_xor(rmax, 1, 16));
      rmax = fmaxf(rmax, __shfl_xor(rmax, 2, 16));
      rmax = fmaxf(rmax, __shfl_xor(rmax, 4, 16));
      rmax = fmaxf(rmax, __shfl_xor(rmax, 8, 16));
      float mnew = fmaxf(mx[r], rmax);
      float alpha = expf(mx[r] - mnew);
      float p0 = (v0 <= -1.0e37f) ? 0.f : expf(v0 - mnew);
      float p1 = (v1 <= -1.0e37f) ? 0.f : expf(v1 - mnew);
      float ps = p0 + p1;
      ps += __shfl_xor(ps, 1, 16); ps += __shfl_xor(ps, 2, 16);
      ps += __shfl_xor(ps, 4, 16); ps += __shfl_xor(ps, 8, 16);
      sm[r] = sm[r] * alpha + ps;
      mx[r] = mnew;
#pragma unroll
      for (int nt = 0; nt < 16; ++nt) acc[nt][r] *= alpha;
      pf[mrow * 32 + col] = (_Float16)p0;
      pf[mrow * 32 + 16 + col] = (_Float16)p1;
    }
    __syncthreads();
    v16h pa;
    {
      int m = lane & 15, k0 = 8 * hi;
      v8h lo = *(const v8h*)&pf[m * 32 + k0];
      v8h up = *(const v8h*)&pf[m * 32 + k0 + 16];
#pragma unroll
      for (int i = 0; i < 8; ++i) { pa[i] = lo[i]; pa[8 + i] = up[i]; }
    }
#pragma unroll
    for (int nt = 0; nt < 16; ++nt)
      acc[nt] = WMMA(pa, BVb[((size_t)nt * 32 + kp) * 32 + lane], acc[nt]);
    __syncthreads();
  }
#pragma unroll
  for (int r = 0; r < 8; ++r) {
    int l = lt * 16 + r + 8 * hi;
    float inv = (sm[r] > 0.f) ? 1.f / sm[r] : 0.f;
#pragma unroll
    for (int nt = 0; nt < 16; ++nt)
      ctx[((size_t)b * 1024 + l) * 256 + nt * 16 + col] = acc[nt][r] * inv;
  }
}

// ---------------------------------------------------------------------------
// Tail
// ---------------------------------------------------------------------------
__global__ void logits_kernel(const float* __restrict__ H, const float* __restrict__ w2,
                              const float* __restrict__ b2, float* __restrict__ lg) {
  int m = blockIdx.x * blockDim.x + threadIdx.x;
  if (m >= 32768) return;
  const float* h = H + (size_t)m * 256;
  float s = b2[0];
  for (int dd = 0; dd < 256; ++dd) s += h[dd] * w2[dd];
  lg[m] = s;
}
__global__ void output_kernel(const float* __restrict__ lg, const float* __restrict__ mask,
                              float* __restrict__ out) {
  int m = blockIdx.x * blockDim.x + threadIdx.x;
  if (m >= 32768) return;
  float x = lg[m];
  float p = 1.f / (1.f + expf(-x));
  out[2 * m] = 1.f - p;
  out[2 * m + 1] = p;
  out[65536 + m] = (x > 0.f) ? 1.f : ((x < 0.f) ? -1.f : 0.f);
  out[98304 + m] = mask[m];
  out[131072 + m] = x;
}

// ---------------------------------------------------------------------------
extern "C" void kernel_launch(void* const* d_in, const int* in_sizes, int n_in,
                              void* d_out, int out_size, void* d_ws, size_t ws_size,
                              hipStream_t stream) {
  const float* noisy   = (const float*)d_in[0];
  const float* maskp   = (const float*)d_in[1];
  const int*   trg     = (const int*)d_in[2];
  const float* se_w0   = (const float*)d_in[3];  const float* se_b0 = (const float*)d_in[4];
  const float* se_w1   = (const float*)d_in[5];  const float* se_b1 = (const float*)d_in[6];
  const float* se_w2   = (const float*)d_in[7];  const float* se_b2 = (const float*)d_in[8];
  const float* w_ih0   = (const float*)d_in[9];  const float* w_hh0 = (const float*)d_in[10];
  const float* b_ih0   = (const float*)d_in[11]; const float* b_hh0 = (const float*)d_in[12];
  const float* w_ih1   = (const float*)d_in[13]; const float* w_hh1 = (const float*)d_in[14];
  const float* b_ih1   = (const float*)d_in[15]; const float* b_hh1 = (const float*)d_in[16];
  const float* attn_w1 = (const float*)d_in[17]; const float* attn_b1 = (const float*)d_in[18];
  const float* comb_w  = (const float*)d_in[19]; const float* comb_b  = (const float*)d_in[20];
  const float* dec_w0  = (const float*)d_in[21]; const float* dec_b0  = (const float*)d_in[22];
  const float* dec_w1  = (const float*)d_in[23]; const float* dec_b1  = (const float*)d_in[24];
  const float* dec_w2  = (const float*)d_in[25]; const float* dec_b2  = (const float*)d_in[26];
  float* out = (float*)d_out;

  char* ws = (char*)d_ws;
  size_t off = 0;
  auto alloc = [&](size_t bytes) { size_t o = off; off += (bytes + 255) & ~(size_t)255; return (void*)(ws + o); };

  _Float16* Bp_se0  = (_Float16*)alloc(1024 * 256 * 2);
  _Float16* Bp_se1  = (_Float16*)alloc(256 * 256 * 2);
  _Float16* Bp_se2  = (_Float16*)alloc(256 * 256 * 2);
  _Float16* Bp_whh0 = (_Float16*)alloc(768 * 256 * 2);
  _Float16* Bp_whh1 = (_Float16*)alloc(768 * 256 * 2);
  _Float16* Bp_wih1 = (_Float16*)alloc(768 * 256 * 2);
  _Float16* Bp_attn = (_Float16*)alloc(256 * 256 * 2);
  _Float16* Bp_comb = (_Float16*)alloc(256 * 768 * 2);
  _Float16* Bp_dec0 = (_Float16*)alloc(256 * 256 * 2);
  _Float16* Bp_dec1 = (_Float16*)alloc(256 * 256 * 2);
  _Float16* Ap_noisy= (_Float16*)alloc(32 * 1024 * 2);
  float* t0   = (float*)alloc(32 * 256 * 4);
  float* t1   = (float*)alloc(32 * 256 * 4);
  float* t2   = (float*)alloc(32 * 256 * 4);
  float* sh   = (float*)alloc(32 * 256 * 4);
  _Float16* Ap_s   = (_Float16*)alloc(32 * 256 * 2);
  _Float16* shrep  = (_Float16*)alloc(32 * 8 * 512 * 2);
  _Float16* o0Ap   = (_Float16*)alloc((size_t)32768 * 256 * 2);
  float*    o1f32  = (float*)   alloc((size_t)32768 * 256 * 4);
  _Float16* Ap_outs= (_Float16*)alloc((size_t)32768 * 256 * 2);
  _Float16* BpModif= (_Float16*)alloc((size_t)32768 * 256 * 2);
  _Float16* BpV    = (_Float16*)alloc((size_t)32768 * 256 * 2);
  float*    ctxf   = (float*)   alloc((size_t)32768 * 256 * 4);
  _Float16* Ap_ctx = (_Float16*)alloc((size_t)32768 * 256 * 2);
  float*    ftmpA  = (float*)   alloc((size_t)32768 * 256 * 4);
  float*    ftmpB  = (float*)   alloc((size_t)32768 * 256 * 4);
  _Float16* ApX1   = (_Float16*)alloc((size_t)32768 * 256 * 2);
  _Float16* ApX2   = (_Float16*)alloc((size_t)32768 * 256 * 2);
  float*    lg     = (float*)alloc(32768 * 4);
  (void)ws_size; (void)in_sizes; (void)n_in; (void)out_size;

  auto packB = [&](const float* W, _Float16* o, int N, int K) {
    pack_b_kernel<<<dim3((N * K + 255) / 256, 1), 256, 0, stream>>>(W, o, N, K);
  };
  // weight packing
  packB(se_w0, Bp_se0, 256, 1024);
  packB(se_w1, Bp_se1, 256, 256);
  packB(se_w2, Bp_se2, 256, 256);
  packB(w_hh0, Bp_whh0, 768, 256);
  packB(w_hh1, Bp_whh1, 768, 256);
  packB(w_ih1, Bp_wih1, 768, 256);
  packB(attn_w1, Bp_attn, 256, 256);
  packB(comb_w, Bp_comb, 256, 768);
  packB(dec_w0, Bp_dec0, 256, 256);
  packB(dec_w1, Bp_dec1, 256, 256);

  // ---- start embed: sh = se(noisy) ----
  pack_a_kernel<<<(32 * 1024 + 255) / 256, 256, 0, stream>>>(noisy, Ap_noisy, 32, 1024);
  gemm_kernel<<<4, 256, 0, stream>>>((const v16h*)Ap_noisy, (const v16h*)Bp_se0, se_b0, t0, 2, 16, 32);
  gelu_kernel<<<32, 256, 0, stream>>>(t0, t1, 8192);
  pack_a_kernel<<<32, 256, 0, stream>>>(t1, Ap_s, 32, 256);
  gemm_kernel<<<4, 256, 0, stream>>>((const v16h*)Ap_s, (const v16h*)Bp_se1, se_b1, t2, 2, 16, 8);
  gelu_kernel<<<32, 256, 0, stream>>>(t2, t1, 8192);
  pack_a_kernel<<<32, 256, 0, stream>>>(t1, Ap_s, 32, 256);
  gemm_kernel<<<4, 256, 0, stream>>>((const v16h*)Ap_s, (const v16h*)Bp_se2, se_b2, t2, 2, 16, 8);
  add_kernel<<<32, 256, 0, stream>>>(t2, t0, sh, 8192);
  shrep_kernel<<<(32 * 8 * 512 + 255) / 256, 256, 0, stream>>>(sh, shrep);

  // ---- GRU layer 0 (scalar x path) -> o0Ap ----
  gru_kernel<<<1, 1024, 0, stream>>>(sh, (const v16h*)Bp_whh0, b_hh0, b_ih0,
                                     trg, w_ih0, nullptr, nullptr, o0Ap, nullptr);
  // ---- GRU layer 1 (x GEMM from o0Ap) -> o1f32 (b-major) ----
  gru_kernel<<<1, 1024, 0, stream>>>(sh, (const v16h*)Bp_whh1, b_hh1, b_ih1,
                                     nullptr, nullptr, (const v16h*)o0Ap,
                                     (const v16h*)Bp_wih1, nullptr, o1f32);

  // ---- attention ----
  pack_a_kernel<<<32768, 256, 0, stream>>>(o1f32, Ap_outs, 32768, 256);
  gemm_kernel<<<4096, 256, 0, stream>>>((const v16h*)Ap_outs, (const v16h*)Bp_attn,
                                        attn_b1, ftmpA, 2048, 16, 8);          // modif
  pack_b_kernel<<<dim3((1024 * 256 + 255) / 256, 32), 256, 0, stream>>>(ftmpA, BpModif, 1024, 256);
  pack_bT_kernel<<<dim3((1024 * 256 + 255) / 256, 32), 256, 0, stream>>>(o1f32, BpV, 256, 1024);
  attn_kernel<<<2048, 32, 0, stream>>>((const v16h*)Ap_outs, (const v16h*)BpModif,
                                       (const v16h*)BpV, ctxf);

  // ---- decoder ----
  pack_a_kernel<<<32768, 256, 0, stream>>>(ctxf, Ap_ctx, 32768, 256);
  gemm_comb_kernel<<<4096, 256, 0, stream>>>((const v16h*)Ap_ctx, (const v16h*)Ap_outs,
                                             (const v16h*)shrep, (const v16h*)Bp_comb,
                                             comb_b, ftmpA);
  tanh_kernel<<<32768, 256, 0, stream>>>(ftmpA, 8388608);
  pack_a_kernel<<<32768, 256, 0, stream>>>(ftmpA, ApX1, 32768, 256);
  gemm_kernel<<<4096, 256, 0, stream>>>((const v16h*)ApX1, (const v16h*)Bp_dec0, dec_b0,
                                        ftmpB, 2048, 16, 8);
  gelu_kernel<<<32768, 256, 0, stream>>>(ftmpB, ftmpB, 8388608);
  pack_a_kernel<<<32768, 256, 0, stream>>>(ftmpB, ApX2, 32768, 256);
  gemm_kernel<<<4096, 256, 0, stream>>>((const v16h*)ApX2, (const v16h*)Bp_dec1, dec_b1,
                                        ftmpA, 2048, 16, 8);
  gelu_kernel<<<32768, 256, 0, stream>>>(ftmpA, ftmpA, 8388608);
  logits_kernel<<<128, 256, 0, stream>>>(ftmpA, dec_w2, dec_b2, lg);
  output_kernel<<<128, 256, 0, stream>>>(lg, maskp, out);
}
